// NeuronCircuitUp_31593779429535
// MI455X (gfx1250) — matrix operand order repaired and checked by reading, not compiled, not run on hardware
//
#include <hip/hip_runtime.h>

// Problem constants (from reference)
#define TOK   8192      // B*S = 4*2048
#define RANK  128
#define DM    1024      // D_MODEL
#define NE    8         // N_OUTPUT (experts)
#define KHH   4         // Householder reflections per token
#define KU    1024      // U inner dim = NE*RANK
#define MT    128       // tokens per block
#define NSUB  8         // 16-row M-subtiles per wave
#define A_STRIDE 136    // LDS row stride in bf16 (128+8): 272B rows, conflict-free b128

typedef __attribute__((ext_vector_type(16))) __bf16 v16bf;
typedef __attribute__((ext_vector_type(8)))  __bf16 v8bf;
typedef __attribute__((ext_vector_type(4)))  __bf16 v4bf;
typedef __attribute__((ext_vector_type(8)))  float  v8f;

union Frag16 { v16bf v; v8bf h[2]; };

#if __has_builtin(__builtin_amdgcn_sched_group_barrier)
#define SCHED_GROUP(mask, size, id) __builtin_amdgcn_sched_group_barrier((mask), (size), (id))
#else
#define SCHED_GROUP(mask, size, id)
#endif

// ---------------------------------------------------------------------------
// Kernel 1: Householder (f32, matches reference math) + build U in bf16:
//   U[t, e*RANK + r] = w[t,e] * xh[t,r]
// One wave32 per token; lane holds x[4l..4l+3]; butterfly reductions.
// ---------------------------------------------------------------------------
__global__ __launch_bounds__(256) void hh_u_kernel(const float* __restrict__ x,
                                                   const float* __restrict__ ow,
                                                   const int* __restrict__ pidx,
                                                   const float* __restrict__ pneur,
                                                   __bf16* __restrict__ U) {
  const int lane  = threadIdx.x & 31;
  const int token = blockIdx.x * 8 + (threadIdx.x >> 5);

  float4 xv = ((const float4*)(x + (size_t)token * RANK))[lane];
  const int* ip = pidx + token * KHH;

#pragma unroll
  for (int k = 0; k < KHH; ++k) {
    const int idx = ip[k];
    float4 vv = ((const float4*)(pneur + (size_t)idx * RANK))[lane];
    float pvv = vv.x * vv.x + vv.y * vv.y + vv.z * vv.z + vv.w * vv.w;
    float pvx = vv.x * xv.x + vv.y * xv.y + vv.z * xv.z + vv.w * xv.w;
#pragma unroll
    for (int off = 16; off >= 1; off >>= 1) {
      pvv += __shfl_xor(pvv, off, 32);
      pvx += __shfl_xor(pvx, off, 32);
    }
    const float coef = 2.0f * pvx / (pvv + 1e-8f);
    xv.x -= coef * vv.x;
    xv.y -= coef * vv.y;
    xv.z -= coef * vv.z;
    xv.w -= coef * vv.w;
  }

  const float* wt = ow + (size_t)token * NE;
  __bf16* urow = U + (size_t)token * KU;
#pragma unroll
  for (int e = 0; e < NE; ++e) {
    const float we = wt[e];
    v4bf u;
    u[0] = (__bf16)(we * xv.x);
    u[1] = (__bf16)(we * xv.y);
    u[2] = (__bf16)(we * xv.z);
    u[3] = (__bf16)(we * xv.w);
    *(v4bf*)(urow + e * RANK + lane * 4) = u;   // 8B per lane, coalesced
  }
}

// ---------------------------------------------------------------------------
// Kernel 2: swizzle W[e,r,d] (f32, viewed as Wflat[KU][DM]) into bf16 fragments.
// Fragment (kt, dt): 32x16 region, K0 = kt*32, N0 = dt*16.
// B layout used here (k order matches A): lane l holds K = K0 + 16*(l>>4) + i (i=0..15), N = N0 + (l&15),
// packed so each lane's 16 elements are 32 contiguous bytes:
//   Bt[frag*512 + l*16 + i],  frag = kt*(DM/16) + dt
// ---------------------------------------------------------------------------
__global__ __launch_bounds__(256) void bprep_kernel(const float* __restrict__ Wn,
                                                    __bf16* __restrict__ Bt) {
  const int gid  = blockIdx.x * 256 + threadIdx.x;   // 0 .. 65535
  const int frag = gid >> 5;                         // 0 .. 2047
  const int l    = gid & 31;
  const int kt   = frag >> 6;                        // KU/32 = 32 values
  const int dt   = frag & 63;                        // DM/16 = 64 values
  const int k0   = kt * 32 + ((l >> 4) << 4);
  const int n    = dt * 16 + (l & 15);

  Frag16 f;
#pragma unroll
  for (int i = 0; i < 16; ++i) {
    f.v[i] = (__bf16)Wn[(size_t)(k0 + i) * DM + n];
  }
  v8bf* dst = (v8bf*)(Bt + (size_t)frag * 512 + l * 16);
  dst[0] = f.h[0];
  dst[1] = f.h[1];
}

// ---------------------------------------------------------------------------
// Kernel 3: out = U(bf16) x Wflat(bf16), f32 accumulate.
// Block (8 waves) = 128 tokens x 128 cols; waves split columns.
// A ktile (128x128 bf16) staged in LDS per 128-K slab. Per K-step:
//   - one B fragment (2x coalesced global_load_b128), reused by 8 WMMAs
//   - 16 ds_load_b128 grouped BEFORE the 8 WMMAs (sched_group_barrier), so
//     the fragments live in distinct registers and the WMMAs issue
//     back-to-back behind a single dscnt wait.
//
// WMMA bf16 operand layouts as used by this kernel:
//   A (16x32): lane l -> M = l&15; elem i -> K = 16*(l>>4)+i   (from LDS rows)
//   B (32x16): lane l -> N = l&15; elem i -> K = 16*(l>>4)+i   (pre-swizzled)
//   C (16x16 f32): VGPR j -> M = j + 8*(l>>4), N = l&15
// ---------------------------------------------------------------------------
__global__ __launch_bounds__(256) void gemm_bf16(const __bf16* __restrict__ U,
                                                 const __bf16* __restrict__ Bt,
                                                 float* __restrict__ out) {
  __shared__ __align__(16) __bf16 a_lds[MT * A_STRIDE];

  const int tid  = threadIdx.x;
  const int t0   = blockIdx.y * MT;
  const int wave = tid >> 5;
  const int lane = tid & 31;
  const int m     = lane & 15;
  const int koff8 = (lane >> 4) << 4;        // 0 or 16 (bf16 elements)
  const int half  = (lane >> 4) << 3;        // C row offset for upper half-wave
  const int dtile = blockIdx.x * 8 + wave;   // global 16-col tile (0..63)
  const int dcol  = dtile * 16 + m;

  v8f acc[NSUB];
#pragma unroll
  for (int s = 0; s < NSUB; ++s)
    acc[s] = (v8f){0.f, 0.f, 0.f, 0.f, 0.f, 0.f, 0.f, 0.f};

#pragma unroll 1
  for (int kt = 0; kt < KU / 128; ++kt) {    // 8 slabs of K=128
    __syncthreads();
    // Stage U[t0..t0+127][kt*128 .. +127] -> LDS.
    // 128 rows x 16 chunks(16B) = 2048 chunks; 256 threads x 8 chunks each.
#pragma unroll
    for (int i = 0; i < 8; ++i) {
      const int idx = tid + i * 256;         // 0..2047
      const int r   = idx >> 4;              // row (16 chunks per row)
      const int c8  = (idx & 15) * 8;        // bf16 column
      *(v8bf*)&a_lds[r * A_STRIDE + c8] =
          *(const v8bf*)&U[(size_t)(t0 + r) * KU + kt * 128 + c8];
    }
    __syncthreads();

#pragma unroll
    for (int ks = 0; ks < 4; ++ks) {         // 4 K-steps of 32 within the slab
      const int kfrag = kt * 4 + ks;         // global K-step index (0..31)
      const v8bf* bp = (const v8bf*)(Bt + ((size_t)kfrag * 64 + dtile) * 512 + lane * 16);
      Frag16 b;
      b.h[0] = bp[0];
      b.h[1] = bp[1];

      // Load all 8 A fragments (16 ds_load_b128) into distinct registers.
      Frag16 a[NSUB];
#pragma unroll
      for (int s = 0; s < NSUB; ++s) {
        const __bf16* ap = &a_lds[(s * 16 + m) * A_STRIDE + ks * 32 + koff8];
        a[s].h[0] = *(const v8bf*)ap;
        a[s].h[1] = *(const v8bf*)(ap + 8);
      }
      // 8 independent WMMAs, back-to-back.
#pragma unroll
      for (int s = 0; s < NSUB; ++s) {
        acc[s] = __builtin_amdgcn_wmma_f32_16x16x32_bf16(
            false, a[s].v, false, b.v, (short)0, acc[s], false, false);
      }
      // Scheduling: 16 DS reads (0x100) first, then 8 WMMA (0x8).
      // VMEM left unconstrained so B loads keep prefetching early.
      SCHED_GROUP(0x100, 16, 0);
      SCHED_GROUP(0x008, 8, 0);
    }
  }

  // Store: subtile s, VGPR j -> token t0 + s*16 + j + half, column dcol.
#pragma unroll
  for (int s = 0; s < NSUB; ++s) {
#pragma unroll
    for (int j = 0; j < 8; ++j) {
      out[(size_t)(t0 + s * 16 + j + half) * DM + dcol] = acc[s][j];
    }
  }
}

// ---------------------------------------------------------------------------
extern "C" void kernel_launch(void* const* d_in, const int* in_sizes, int n_in,
                              void* d_out, int out_size, void* d_ws, size_t ws_size,
                              hipStream_t stream) {
  const float* x     = (const float*)d_in[0];   // [B,S,RANK]    f32
  const float* ow    = (const float*)d_in[1];   // [B,S,NE]      f32
  const int*   pidx  = (const int*)d_in[2];     // [B,S,K]       i32
  const float* pneur = (const float*)d_in[3];   // [NP,RANK]     f32
  const float* oneur = (const float*)d_in[4];   // [NE,RANK,DM]  f32
  float* out = (float*)d_out;                   // [B,S,DM]      f32

  // Workspace: U bf16 (16 MB) | Bt bf16 (2 MB)
  __bf16* U  = (__bf16*)d_ws;
  __bf16* Bt = (__bf16*)((char*)d_ws + (size_t)TOK * KU * sizeof(__bf16));

  hh_u_kernel<<<TOK / 8, 256, 0, stream>>>(x, ow, pidx, pneur, U);
  bprep_kernel<<<(KU / 32) * (DM / 16) * 32 / 256, 256, 0, stream>>>(oneur, Bt);

  dim3 grid(DM / 128, TOK / MT);                // 8 x 64 = 512 blocks
  gemm_bf16<<<grid, 256, 0, stream>>>(U, Bt, out);
}
